// ProbAttention_30039001268326
// MI455X (gfx1250) — matrix operand order repaired and checked by reading, not compiled, not run on hardware
//
#include <hip/hip_runtime.h>
#include <hip/hip_bf16.h>
#include <math.h>

typedef __attribute__((ext_vector_type(16))) _Float16 v16h;
typedef __attribute__((ext_vector_type(8)))  _Float16 v8h;
typedef __attribute__((ext_vector_type(8)))  float    v8f;

#define BB 16
#define NN 1024
#define DMAX 512
#define DLEN 256
#define DMOD 128
#define NCLS 10
#define SK   140

__device__ inline v8f wmma_f16(v16h a, v16h b, v8f c) {
  // D = A(16x32 f16) * B(32x16 f16) + C(16x16 f32)
  return __builtin_amdgcn_wmma_f32_16x16x32_f16(
      /*neg_a=*/false, a, /*neg_b=*/false, b,
      /*c_mod=*/(short)0, c, /*reuse_a=*/false, /*reuse_b=*/false);
}

// combine two contiguous 8-half runs into one A/B fragment register set
__device__ inline v16h cat8(v8h lo, v8h hi) {
  return __builtin_shufflevector(lo, hi, 0, 1, 2, 3, 4, 5, 6, 7,
                                          8, 9, 10, 11, 12, 13, 14, 15);
}

// ---------------------------------------------------------------------------
// Generic C[M,N] = A[M,K] @ B[K,N] (+bias[N]), f16 WMMA compute, f32 accum.
// Block = 256 threads = 8 waves. Wave w owns a 16-row tile (blockIdx.x*8+w)
// and a 16x64 output strip (4 accumulators): one A fragment feeds 4 WMMAs.
// B 32x64 tile staged TRANSPOSED in LDS (btT[n][k]) so a lane's B-fragment
// (fixed n, k = 16*g..16*g+15) is 16 contiguous halves -> 2x ds_load_b128.
// A-frag (16-bit A 16x32): lane l: m=l&15, kbase=16*(l>>4);
//   halves i -> K=kbase+i (contiguous, same order as B-frag).
// B-frag: lane l: n=l&15, halves i -> K=16*(l>>4)+i.
// C/D f32: reg r, lane l -> M = r + 8*(l>>4), N = l&15 (per ISA table).
// Requires: M%128==0, N%64==0, K%32==0  (16384/{256,128}/{512,256} all OK).
// ---------------------------------------------------------------------------
__global__ void __launch_bounds__(256)
wmma_gemm_kernel(const float* __restrict__ A, const float* __restrict__ Bm,
                 const float* __restrict__ bias, float* __restrict__ C,
                 int M, int N, int K)
{
  __shared__ _Float16 btT[64][32];                // transposed B tile, 4 KB
  const int tid  = threadIdx.x;
  const int wave = tid >> 5, lane = tid & 31;
  const int g = lane >> 4, hl = lane & 15;
  const int m0 = (blockIdx.x * 8 + wave) * 16;
  const int n0 = blockIdx.y * 64;
  const float* arow = A + (size_t)(m0 + hl) * K;
  const int sn = tid & 63;                        // staging: column 0..63
  const int sk = (tid >> 6) * 8;                  // staging: k-run {0,8,16,24}
  v8f acc0 = {}, acc1 = {}, acc2 = {}, acc3 = {};
  for (int k0 = 0; k0 < K; k0 += 32) {
    // ---- stage B 32x64 tile -> LDS transposed (8 elems/thread) ----
    {
      const float* src = Bm + (size_t)(k0 + sk) * N + n0 + sn;  // coalesced in sn
      v8h col;
#pragma unroll
      for (int j = 0; j < 8; ++j) col[j] = (_Float16)src[(size_t)j * N];
      *reinterpret_cast<v8h*>(&btT[sn][sk]) = col;              // ds_store_b128
    }
    __syncthreads();
    // ---- A fragment: two 8-float runs, b128 global loads + pk cvt ----
    v16h af;
    {
      const float4* ap4 = reinterpret_cast<const float4*>(arow + k0 + 16 * g);
      float4 x0 = ap4[0], x1 = ap4[1];            // K = 16g+0..7
      float4 x2 = ap4[2], x3 = ap4[3];            // K = 16g+8..15
      af[0]=(_Float16)x0.x; af[1]=(_Float16)x0.y; af[2]=(_Float16)x0.z; af[3]=(_Float16)x0.w;
      af[4]=(_Float16)x1.x; af[5]=(_Float16)x1.y; af[6]=(_Float16)x1.z; af[7]=(_Float16)x1.w;
      af[8]=(_Float16)x2.x; af[9]=(_Float16)x2.y; af[10]=(_Float16)x2.z; af[11]=(_Float16)x2.w;
      af[12]=(_Float16)x3.x; af[13]=(_Float16)x3.y; af[14]=(_Float16)x3.z; af[15]=(_Float16)x3.w;
    }
    // ---- 4 WMMAs sharing one A fragment; B frags are contiguous LDS ----
    v16h bf0 = *reinterpret_cast<const v16h*>(&btT[ 0 + hl][16 * g]);
    v16h bf1 = *reinterpret_cast<const v16h*>(&btT[16 + hl][16 * g]);
    v16h bf2 = *reinterpret_cast<const v16h*>(&btT[32 + hl][16 * g]);
    v16h bf3 = *reinterpret_cast<const v16h*>(&btT[48 + hl][16 * g]);
    acc0 = wmma_f16(af, bf0, acc0);
    acc1 = wmma_f16(af, bf1, acc1);
    acc2 = wmma_f16(af, bf2, acc2);
    acc3 = wmma_f16(af, bf3, acc3);
    __syncthreads();
  }
  float bv0 = bias ? bias[n0 +  0 + hl] : 0.0f;
  float bv1 = bias ? bias[n0 + 16 + hl] : 0.0f;
  float bv2 = bias ? bias[n0 + 32 + hl] : 0.0f;
  float bv3 = bias ? bias[n0 + 48 + hl] : 0.0f;
#pragma unroll
  for (int r = 0; r < 8; ++r) {
    float* crow = C + (size_t)(m0 + r + 8 * g) * N + n0 + hl;
    crow[ 0] = acc0[r] + bv0;
    crow[16] = acc1[r] + bv1;
    crow[32] = acc2[r] + bv2;
    crow[48] = acc3[r] + bv3;
  }
}

// ---------------------------------------------------------------------------
// M[b,l] = max_s(q_l . k_{idx[l,s]}) - sum_s(...)/1024 over 140 samples.
// One wave per (b,l); q cached in LDS; gather pattern -> VALU dots.
// ---------------------------------------------------------------------------
__global__ void __launch_bounds__(128)
prob_m_kernel(const float* __restrict__ Qb, const float* __restrict__ Kb,
              const int* __restrict__ idx, float* __restrict__ Mv)
{
  __shared__ float qs[4][DMOD];
  __shared__ float redm[4][32], reds[4][32];
  const int tid = threadIdx.x, wave = tid >> 5, lane = tid & 31;
  const int p = blockIdx.x * 4 + wave;          // 0..16383
  const int b = p >> 10, l = p & 1023;
  for (int d = lane; d < DMOD; d += 32)
    qs[wave][d] = Qb[((size_t)b * NN + l) * DMOD + d];
  __syncthreads();
  float mx = -INFINITY, sm = 0.f;
  for (int s = lane; s < SK; s += 32) {
    const float4* kr = reinterpret_cast<const float4*>(
        Kb + ((size_t)b * NN + idx[l * SK + s]) * DMOD);
    float dot = 0.f;
#pragma unroll 8
    for (int d4 = 0; d4 < DMOD / 4; ++d4) {
      float4 k4 = kr[d4];
      dot = fmaf(qs[wave][d4 * 4 + 0], k4.x, dot);
      dot = fmaf(qs[wave][d4 * 4 + 1], k4.y, dot);
      dot = fmaf(qs[wave][d4 * 4 + 2], k4.z, dot);
      dot = fmaf(qs[wave][d4 * 4 + 3], k4.w, dot);
    }
    mx = fmaxf(mx, dot);  sm += dot;
  }
  redm[wave][lane] = mx;  reds[wave][lane] = sm;
  __syncthreads();
  if (lane == 0) {
    float M = -INFINITY, S = 0.f;
    for (int i = 0; i < 32; ++i) { M = fmaxf(M, redm[wave][i]); S += reds[wave][i]; }
    Mv[p] = M - S * (1.0f / 1024.0f);
  }
}

// ---------------------------------------------------------------------------
// Per-batch top-140 indices of M[b,:1024]; ties -> lowest index (JAX rule).
// ---------------------------------------------------------------------------
__global__ void __launch_bounds__(256)
topk_kernel(const float* __restrict__ Mv, int* __restrict__ Mtop)
{
  __shared__ float vals[NN];
  __shared__ float rv[256];
  __shared__ int   ra[256];
  const int b = blockIdx.x, tid = threadIdx.x;
  for (int i = tid; i < NN; i += 256) vals[i] = Mv[b * NN + i];
  __syncthreads();
  for (int t = 0; t < SK; ++t) {
    float bm = -INFINITY; int ba = 0x7fffffff;
#pragma unroll
    for (int j = 0; j < 4; ++j) {
      int i = tid * 4 + j;
      float v = vals[i];
      if (v > bm) { bm = v; ba = i; }      // ascending scan keeps lowest index
    }
    rv[tid] = bm; ra[tid] = ba;
    __syncthreads();
    for (int off = 128; off > 0; off >>= 1) {
      if (tid < off) {
        if (rv[tid + off] > rv[tid] ||
            (rv[tid + off] == rv[tid] && ra[tid + off] < ra[tid])) {
          rv[tid] = rv[tid + off]; ra[tid] = ra[tid + off];
        }
      }
      __syncthreads();
    }
    if (tid == 0) { Mtop[b * SK + t] = ra[0]; vals[ra[0]] = -INFINITY; }
    __syncthreads();
  }
}

__global__ void __launch_bounds__(128)
meanv_kernel(const float* __restrict__ Vb, float* __restrict__ meanV)
{
  const int b = blockIdx.x, d = threadIdx.x;
  float s = 0.f;
  for (int n = 0; n < NN; ++n) s += Vb[((size_t)b * NN + n) * DMOD + d];
  meanV[b * DMOD + d] = s * (1.0f / 1024.0f);
}

// OutPre[b,n,d] = mean(V)[b,d] + XA[b,n,d]  (ctx base everywhere)
__global__ void __launch_bounds__(256)
fill_kernel(const float* __restrict__ meanV, const float* __restrict__ XA,
            float* __restrict__ OutPre)
{
  size_t i = (size_t)blockIdx.x * 256 + threadIdx.x;  // 16*1024*128 total
  int d = (int)(i & 127);
  int b = (int)(i >> 17);
  OutPre[i] = meanV[b * DMOD + d] + XA[i];
}

// ---------------------------------------------------------------------------
// Dense attention for selected queries. Block = (utile of 16 rows, batch).
// scores (WMMA, f32->f16 LDS) -> softmax (LDS) -> attn@V (WMMA) -> scatter.
// A fragments come from LDS as two contiguous v8h runs -> ds_load_b128.
// ---------------------------------------------------------------------------
__global__ void __launch_bounds__(256)
attn_kernel(const float* __restrict__ Qb, const float* __restrict__ Kb,
            const float* __restrict__ Vb, const float* __restrict__ XA,
            const int* __restrict__ Mtop, float* __restrict__ OutPre)
{
  __shared__ _Float16 qh[16][DMOD];    // 4 KB: selected Q rows (f16)
  __shared__ _Float16 sc[16][NN];      // 32 KB: scores / probs (f16)
  __shared__ float    red[16][17];
  __shared__ int      ridx[16];
  const int b = blockIdx.y, ut = blockIdx.x;   // ut in 0..8 covers 144 >= 140
  const int tid = threadIdx.x, wave = tid >> 5, lane = tid & 31;
  const int g = lane >> 4, hl = lane & 15;

  if (tid < 16) {
    int u = ut * 16 + tid;
    ridx[tid] = (u < SK) ? Mtop[b * SK + u] : -1;
  }
  __syncthreads();
  {  // gather 16 selected Q rows -> f16 LDS
    int r = tid >> 4, d0 = (tid & 15) * 8;
    int qrow = ridx[r] < 0 ? 0 : ridx[r];
    const float4* src = reinterpret_cast<const float4*>(
        Qb + ((size_t)b * NN + qrow) * DMOD + d0);
    float4 q0 = src[0], q1 = src[1];
    v8h q;
    q[0]=(_Float16)q0.x; q[1]=(_Float16)q0.y; q[2]=(_Float16)q0.z; q[3]=(_Float16)q0.w;
    q[4]=(_Float16)q1.x; q[5]=(_Float16)q1.y; q[6]=(_Float16)q1.z; q[7]=(_Float16)q1.w;
    *reinterpret_cast<v8h*>(&qh[r][d0]) = q;     // ds_store_b128
  }
  __syncthreads();

  const float scale = 0.0883883476483184f;  // 1/sqrt(128)
  for (int kt = wave * 8; kt < wave * 8 + 8; ++kt) {   // 64 key tiles / 8 waves
    v8f acc = {};
#pragma unroll
    for (int c = 0; c < 4; ++c) {
      int d0 = c * 32;
      const v8h* q8 = reinterpret_cast<const v8h*>(&qh[hl][d0 + 16 * g]);
      v16h af = cat8(q8[0], q8[1]);       // K=16g+0..15
      const float4* kr = reinterpret_cast<const float4*>(
          Kb + ((size_t)b * NN + kt * 16 + hl) * DMOD + d0 + 16 * g);
      float4 k0 = kr[0], k1 = kr[1], k2 = kr[2], k3 = kr[3];
      v16h bf;
      bf[0]=(_Float16)k0.x;  bf[1]=(_Float16)k0.y;  bf[2]=(_Float16)k0.z;  bf[3]=(_Float16)k0.w;
      bf[4]=(_Float16)k1.x;  bf[5]=(_Float16)k1.y;  bf[6]=(_Float16)k1.z;  bf[7]=(_Float16)k1.w;
      bf[8]=(_Float16)k2.x;  bf[9]=(_Float16)k2.y;  bf[10]=(_Float16)k2.z; bf[11]=(_Float16)k2.w;
      bf[12]=(_Float16)k3.x; bf[13]=(_Float16)k3.y; bf[14]=(_Float16)k3.z; bf[15]=(_Float16)k3.w;
      acc = wmma_f16(af, bf, acc);
    }
#pragma unroll
    for (int r = 0; r < 8; ++r)
      sc[r + 8 * g][kt * 16 + hl] = (_Float16)(acc[r] * scale);
  }
  __syncthreads();

  {  // softmax over 1024 keys per row; 16 threads per row, 64 cols each
    int r = tid >> 4, seg = tid & 15, c0 = seg * 64;
    float lm = -INFINITY;
    for (int c = 0; c < 64; ++c) lm = fmaxf(lm, (float)sc[r][c0 + c]);
    red[r][seg] = lm;
    __syncthreads();
    if (seg == 0) { float m = -INFINITY; for (int i = 0; i < 16; ++i) m = fmaxf(m, red[r][i]); red[r][16] = m; }
    __syncthreads();
    float rm = red[r][16];
    float ls = 0.f;
    for (int c = 0; c < 64; ++c) {
      float e = __expf((float)sc[r][c0 + c] - rm);
      sc[r][c0 + c] = (_Float16)e;
      ls += e;
    }
    red[r][seg] = ls;
    __syncthreads();
    if (seg == 0) { float s = 0.f; for (int i = 0; i < 16; ++i) s += red[r][i]; red[r][16] = 1.0f / s; }
    __syncthreads();
    float inv = red[r][16];
    for (int c = 0; c < 64; ++c) sc[r][c0 + c] = (_Float16)((float)sc[r][c0 + c] * inv);
  }
  __syncthreads();

  {  // upd = attn[16,1024] @ V[1024,128]; wave owns 16-col tile
    const int ntile = wave;
    v8f acc = {};
    for (int c = 0; c < 32; ++c) {
      int k0 = c * 32;
      const v8h* a8 = reinterpret_cast<const v8h*>(&sc[hl][k0 + 16 * g]);
      v16h af = cat8(a8[0], a8[1]);       // contiguous runs -> ds_load_b128
      v16h bf;
      const float* vb = Vb + ((size_t)b * NN + k0 + 16 * g) * DMOD + ntile * 16 + hl;
#pragma unroll
      for (int i = 0; i < 16; ++i) bf[i] = (_Float16)vb[(size_t)i * DMOD];
      acc = wmma_f16(af, bf, acc);
    }
#pragma unroll
    for (int r = 0; r < 8; ++r) {
      int m = r + 8 * g;
      int pos = ridx[m];
      if (pos >= 0) {
        size_t off = ((size_t)b * NN + pos) * DMOD + ntile * 16 + hl;
        OutPre[off] = acc[r] + XA[off];   // overwrite ctx-base row with upd+XA
      }
    }
  }
}

// ---------------------------------------------------------------------------
// out[16,10] = OutPre[16,131072] @ W_final + b_final, deterministic 2-stage.
// ---------------------------------------------------------------------------
__global__ void __launch_bounds__(160)
final_partial(const float* __restrict__ OutPre, const float* __restrict__ Wf,
              float* __restrict__ part)
{
  const int j = blockIdx.x, t = threadIdx.x;  // 128 K-slabs, 160 (m,c) pairs
  const int m = t / NCLS, c = t % NCLS;
  const size_t k0 = (size_t)j * 1024;
  const float* xr = OutPre + (size_t)m * (NN * DMOD) + k0;
  const float* wc = Wf + k0 * NCLS + c;
  float s = 0.f;
  for (int k = 0; k < 1024; ++k) s = fmaf(xr[k], wc[(size_t)k * NCLS], s);
  part[(size_t)j * 160 + t] = s;
}

__global__ void __launch_bounds__(160)
final_reduce(const float* __restrict__ part, const float* __restrict__ bias,
             float* __restrict__ out)
{
  const int t = threadIdx.x;
  float s = bias[t % NCLS];
  for (int j = 0; j < 128; ++j) s += part[(size_t)j * 160 + t];  // fixed order
  out[t] = s;
}

// ---------------------------------------------------------------------------
extern "C" void kernel_launch(void* const* d_in, const int* in_sizes, int n_in,
                              void* d_out, int out_size, void* d_ws, size_t ws_size,
                              hipStream_t stream) {
  const float* emb     = (const float*)d_in[0];
  const int*   idx     = (const int*)  d_in[1];
  const float* W_lin   = (const float*)d_in[2];
  const float* b_lin   = (const float*)d_in[3];
  const float* W_q     = (const float*)d_in[4];
  const float* W_k     = (const float*)d_in[5];
  const float* W_v     = (const float*)d_in[6];
  const float* W_add   = (const float*)d_in[7];
  const float* b_add   = (const float*)d_in[8];
  const float* W_final = (const float*)d_in[9];
  const float* b_final = (const float*)d_in[10];
  float* out = (float*)d_out;

  // workspace layout (floats)
  float* ws     = (float*)d_ws;
  float* X      = ws;                                   // 16384*256
  float* Qb     = X      + (size_t)16384 * 256;         // 16384*128
  float* Kb     = Qb     + (size_t)16384 * 128;
  float* Vb     = Kb     + (size_t)16384 * 128;
  float* XA     = Vb     + (size_t)16384 * 128;
  float* Mv     = XA     + (size_t)16384 * 128;         // 16384
  int*   Mtop   = (int*)(Mv + 16384);                   // 16*140
  float* meanV  = (float*)(Mtop + BB * SK);             // 16*128
  float* OutPre = meanV  + BB * DMOD;                   // 16384*128
  float* part   = OutPre + (size_t)16384 * 128;         // 128*160

  // 1) X = Emb @ W_lin + b_lin          (M=16384, N=256, K=512)
  wmma_gemm_kernel<<<dim3(128, 4), 256, 0, stream>>>(emb, W_lin, b_lin, X, 16384, DLEN, DMAX);
  // 2) Q, K, V, XA from X               (M=16384, N=128, K=256)
  wmma_gemm_kernel<<<dim3(128, 2), 256, 0, stream>>>(X, W_q,   nullptr, Qb, 16384, DMOD, DLEN);
  wmma_gemm_kernel<<<dim3(128, 2), 256, 0, stream>>>(X, W_k,   nullptr, Kb, 16384, DMOD, DLEN);
  wmma_gemm_kernel<<<dim3(128, 2), 256, 0, stream>>>(X, W_v,   nullptr, Vb, 16384, DMOD, DLEN);
  wmma_gemm_kernel<<<dim3(128, 2), 256, 0, stream>>>(X, W_add, b_add,   XA, 16384, DMOD, DLEN);
  // 3) sparsity measure M
  prob_m_kernel<<<4096, 128, 0, stream>>>(Qb, Kb, idx, Mv);
  // 4) per-batch top-140
  topk_kernel<<<BB, 256, 0, stream>>>(Mv, Mtop);
  // 5) ctx base = mean(V); OutPre = ctx_base + XA
  meanv_kernel<<<BB, 128, 0, stream>>>(Vb, meanV);
  fill_kernel<<<8192, 256, 0, stream>>>(meanV, XA, OutPre);
  // 6) dense attention on selected queries, scatter upd+XA
  attn_kernel<<<dim3(9, BB), 256, 0, stream>>>(Qb, Kb, Vb, XA, Mtop, OutPre);
  // 7) final classifier GEMM (deterministic 2-stage reduction)
  final_partial<<<128, 160, 0, stream>>>(OutPre, W_final, part);
  final_reduce<<<1, 160, 0, stream>>>(part, b_final, out);
}
